// MI_NCE_1778116461039
// MI455X (gfx1250) — matrix operand order repaired and checked by reading, not compiled, not run on hardware
//
#include <hip/hip_runtime.h>
#include <math.h>

// ---------------------------------------------------------------------------
// MI455X (gfx1250): 1-layer GCN x3 + projection + NT-Xent contrastive loss.
// GEMMs via v_wmma_f32_16x16x32_bf16; adjacency streamed with the Tensor
// Data Mover (4-deep TENSORcnt pipeline); bf16 conversion via convertvector.
// ---------------------------------------------------------------------------

typedef __attribute__((ext_vector_type(16))) __bf16       v16bf;
typedef __attribute__((ext_vector_type(16))) float        v16f;
typedef __attribute__((ext_vector_type(8)))  float        v8f;
typedef __attribute__((ext_vector_type(8)))  __bf16       v8bf;
typedef __attribute__((ext_vector_type(4)))  unsigned int u32x4;
typedef __attribute__((ext_vector_type(4)))  unsigned int v4u;
typedef __attribute__((ext_vector_type(8)))  int          v8i;
typedef __attribute__((ext_vector_type(4)))  int          v4i;
typedef __attribute__((ext_vector_type(2)))  float        f32x2;
typedef __attribute__((ext_vector_type(2)))  __bf16       bf16x2;

#define NN   8192
#define FF   512
#define HH   128
#define TAUF 0.5f
#define EPSF 1e-8f

enum { EPI_NONE = 0, EPI_BIAS = 1, EPI_PRELU = 2, EPI_ELU = 3 };

#if defined(__AMDGCN__) && __has_builtin(__builtin_amdgcn_tensor_load_to_lds)
#define HAVE_TDM 1
#else
#define HAVE_TDM 0
#endif

// ---- f32 -> bf16 via vector truncation (backend picks v_cvt_pk_bf16_f32) --
__device__ __forceinline__ unsigned int pack_bf2(float lo, float hi) {
    f32x2 f; f[0] = lo; f[1] = hi;
    bf16x2 b = __builtin_convertvector(f, bf16x2);
    union { bf16x2 v; unsigned int u; } c; c.v = b;
    return c.u;
}

// A fragment (ISA 16-bit A 16x32 layout): seg0 = [0..8), seg1 = [16..24)
__device__ __forceinline__ v16bf make_afrag_f32(const float* ap) {
    union { v16f f; float4 q[4]; } t;
    t.q[0] = *(const float4*)(ap + 0);
    t.q[1] = *(const float4*)(ap + 4);
    t.q[2] = *(const float4*)(ap + 16);
    t.q[3] = *(const float4*)(ap + 20);
    return __builtin_convertvector(t.f, v16bf);
}

// ---- cooperative pack of B[k0..k0+32) x 128 (row-major f32) into LDS in the
//      ISA B-fragment layout: [tile][lane = K-half*16 | col&15][vgpr] --------
__device__ __forceinline__ void fill_packB(unsigned int (*pb)[32][8],
                                           const float* __restrict__ B,
                                           int k0, int tid) {
    const int n   = tid & 127;
    const int kp0 = (tid >> 7) << 3;
    const int t   = n >> 4;
    const int l   = n & 15;
    #pragma unroll
    for (int j = 0; j < 8; ++j) {
        int kp = kp0 + j;                // k' in 0..15, k = 2k'
        const float* bp = B + (size_t)(k0 + 2 * kp) * 128 + n;
        pb[t][(((kp >> 2) & 1) << 4) | l][(kp & 3) | ((kp >> 3) << 2)] = pack_bf2(bp[0], bp[128]);
    }
}

// ---- TDM: load [128 rows x 32 cols] f32 tile into LDS, 4-dword pad per
//      32-dword row (LDS row stride 36 dwords -> conflict-free b128 reads) ---
#define AW 36
#define TDEPTH 4
#define TILE_BYTES (128 * AW * 4)

__device__ __forceinline__ void tdm_load_tile(const float* gsrc,
                                              unsigned lds_byte_off, int lda) {
#if HAVE_TDM
    unsigned long long ga = (unsigned long long)(uintptr_t)gsrc;
    v4u g0;
    g0[0] = 1u;                                            // count=1
    g0[1] = lds_byte_off;
    g0[2] = (unsigned)(ga & 0xFFFFFFFFull);
    g0[3] = (unsigned)((ga >> 32) & 0x1FFFFFFull) | (2u << 30);  // type=2
    v8i g1;
    // data_size=4B(2), pad_enable, pad_interval=32dw(4), pad_amount=4dw(3)
    g1[0] = (int)((2u << 16) | (1u << 20) | (4u << 22) | (3u << 25));
    g1[1] = 0;                                   // tensor_dim0 = 2^30 (lo16=0)
    g1[2] = 0x4000;                              // dim0 hi16, dim1 lo16
    g1[3] = (int)(0x4000u | (32u << 16));        // dim1 hi16, tile_dim0 = 32
    g1[4] = 128;                                 // tile_dim1=128, tile_dim2=0
    g1[5] = lda;                                 // tensor_dim0_stride
    g1[6] = 0;
    g1[7] = 0;
    v4i z4 = {};
#if defined(__clang_major__) && (__clang_major__ >= 23)
    v8i z8 = {};
    __builtin_amdgcn_tensor_load_to_lds(g0, g1, z4, z4, z8, 0);
#else
    __builtin_amdgcn_tensor_load_to_lds(g0, g1, z4, z4, 0);
#endif
#endif
}

// ---------------------------------------------------------------------------
// GEMM: C[M x 128] = A[M x K] @ B[K x 128]  (ldb = ldc = 128, K % 128 == 0)
// 256 threads (8 waves) per block, 128 rows; A slabs streamed by a 4-deep
// TDM ring; B slabs packed to bf16 fragments in double-buffered LDS.
// ---------------------------------------------------------------------------
__global__ __launch_bounds__(256) void wmma_gemm128(
    const float* __restrict__ A, const float* __restrict__ B,
    float* __restrict__ C, int K, int lda,
    const float* __restrict__ bias, const float* __restrict__ slopep, int mode)
{
    __shared__ __align__(16) float ldsA[TDEPTH][128][AW];  // 72 KB ring
    __shared__ unsigned int packB[2][8][32][8];            // 16 KB

    const int tid  = threadIdx.x;
    const int wave = tid >> 5;
    const int lane = tid & 31;
    const int hlf  = lane >> 4;
    const int l15  = lane & 15;
    const int rowBlk = blockIdx.x * 128;

    v8f acc[8] = {};

#if HAVE_TDM
    const unsigned abase = (unsigned)(uintptr_t)&ldsA[0][0][0];
    if (wave == 0) {
        #pragma unroll
        for (int i = 0; i < TDEPTH; ++i)     // K >= 128 always -> 4 tiles exist
            tdm_load_tile(A + (size_t)rowBlk * lda + 32 * i,
                          abase + i * TILE_BYTES, lda);
    }
#endif
    fill_packB(packB[0], B, 0, tid);

    for (int k0 = 0; k0 < K; k0 += 32) {
        const int cur  = (k0 >> 5) & (TDEPTH - 1);
        const int bcur = (k0 >> 5) & 1;
#if HAVE_TDM
        if (wave == 0) {                      // wait for tile (k0/32) complete
            int rem = ((K - k0) >> 5) - 1;    // tiles still ahead of current
            if (rem >= 3)      __builtin_amdgcn_s_wait_tensorcnt(3);
            else if (rem == 2) __builtin_amdgcn_s_wait_tensorcnt(2);
            else if (rem == 1) __builtin_amdgcn_s_wait_tensorcnt(1);
            else               __builtin_amdgcn_s_wait_tensorcnt(0);
        }
#else
        {   // fallback: cooperative global -> LDS stage of the A slab
            int r = tid >> 1, seg = tid & 1;
            const float4* g = (const float4*)(A + (size_t)(rowBlk + r) * lda
                                              + k0 + seg * 16);
            float4* s = (float4*)&ldsA[cur][r][seg * 16];
            #pragma unroll
            for (int q = 0; q < 4; ++q) s[q] = g[q];
        }
#endif
        __syncthreads();   // ldsA[cur] + packB[bcur] ready for all waves

        if (k0 + 32 < K) fill_packB(packB[bcur ^ 1], B, k0 + 32, tid);

        v16bf af = make_afrag_f32(&ldsA[cur][wave * 16 + l15][hlf * 8]);

        u32x4 b0[8], b1[8];
        #pragma unroll
        for (int t = 0; t < 8; ++t) {
            b0[t] = *(const u32x4*)&packB[bcur][t][lane][0];
            b1[t] = *(const u32x4*)&packB[bcur][t][lane][4];
        }
        #pragma unroll
        for (int t = 0; t < 8; ++t) {
            union { v16bf v; u32x4 p[2]; } bf;
            bf.p[0] = b0[t]; bf.p[1] = b1[t];
            acc[t] = __builtin_amdgcn_wmma_f32_16x16x32_bf16(
                false, af, false, bf.v, (short)0, acc[t], false, false);
        }
        __syncthreads();   // consumption of ldsA[cur] / packB done
#if HAVE_TDM
        if (wave == 0 && k0 + 32 * TDEPTH < K)
            tdm_load_tile(A + (size_t)rowBlk * lda + (k0 + 32 * TDEPTH),
                          abase + cur * TILE_BYTES, lda);
#endif
    }

    // ---- epilogue: bias / PReLU / ELU ------------------------------------
    float slope = (mode == EPI_PRELU) ? slopep[0] : 0.f;
    #pragma unroll
    for (int t = 0; t < 8; ++t) {
        int n = t * 16 + l15;
        float bsv = (mode != EPI_NONE) ? bias[n] : 0.f;
        #pragma unroll
        for (int v = 0; v < 8; ++v) {
            int m = rowBlk + wave * 16 + (hlf << 3) + v;
            float x = acc[t][v] + bsv;
            if (mode == EPI_PRELU)      x = (x >= 0.f) ? x : slope * x;
            else if (mode == EPI_ELU)   x = (x > 0.f)  ? x : (__expf(x) - 1.f);
            C[(size_t)m * 128 + n] = x;
        }
    }
}

// ---------------------------------------------------------------------------
// f32 [NN x 128] -> bf16 copy (8 elements/thread, b128 stores)
// ---------------------------------------------------------------------------
__global__ __launch_bounds__(256) void cvt_bf16(
    const float* __restrict__ src, unsigned short* __restrict__ dst)
{
    size_t i = ((size_t)blockIdx.x * 256 + threadIdx.x) * 8;
    v8f x = *(const v8f*)(src + i);
    v8bf y = __builtin_convertvector(x, v8bf);
    union { v8bf v; u32x4 p; } c; c.v = y;
    *(u32x4*)(dst + i) = c.p;
}

// ---------------------------------------------------------------------------
// Row L2 norms: one row per wave, float4 per lane, shuffle reduce
// ---------------------------------------------------------------------------
__global__ __launch_bounds__(256) void norm128(
    const float* __restrict__ Z, float* __restrict__ out)
{
    int wave = threadIdx.x >> 5, lane = threadIdx.x & 31;
    int row = blockIdx.x * 8 + wave;
    float4 v = ((const float4*)(Z + (size_t)row * 128))[lane];
    float s = v.x * v.x + v.y * v.y + v.z * v.z + v.w * v.w;
    #pragma unroll
    for (int off = 16; off > 0; off >>= 1) s += __shfl_xor(s, off, 32);
    if (lane == 0) out[row] = sqrtf(s);
}

__global__ __launch_bounds__(256) void diag128(
    const float* __restrict__ Z1, const float* __restrict__ Z2,
    const float* __restrict__ n1, const float* __restrict__ n2,
    float* __restrict__ d)
{
    int wave = threadIdx.x >> 5, lane = threadIdx.x & 31;
    int row = blockIdx.x * 8 + wave;
    float4 x = ((const float4*)(Z1 + (size_t)row * 128))[lane];
    float4 y = ((const float4*)(Z2 + (size_t)row * 128))[lane];
    float s = x.x * y.x + x.y * y.y + x.z * y.z + x.w * y.w;
    #pragma unroll
    for (int off = 16; off > 0; off >>= 1) s += __shfl_xor(s, off, 32);
    if (lane == 0) d[row] = __expf(s / (n1[row] * n2[row]) * (1.0f / TAUF));
}

// ---------------------------------------------------------------------------
// NT-Xent tiles on pre-converted bf16 Z: pure b128 loads + WMMA, no cvt VALU.
// A frag: row mRow, elems [k0+hlf*8..+8) and [k0+16+hlf*8..+8).
// B frag: row (colB+t*16+l15), elems [k0+hlf*8..+8) and [k0+16+hlf*8..+8).
// ---------------------------------------------------------------------------
__global__ __launch_bounds__(256) void nt_xent_tiles(
    const unsigned short* __restrict__ Z1b, const unsigned short* __restrict__ Z2b,
    const float* __restrict__ n1, const float* __restrict__ n2,
    float* __restrict__ rowsum, float* __restrict__ colsum)
{
    __shared__ float rn[128], cn[128], colpart[128];

    const int tid  = threadIdx.x;
    const int wave = tid >> 5;
    const int lane = tid & 31;
    const int hlf  = lane >> 4;
    const int l15  = lane & 15;
    const int rowB = blockIdx.y * 128;
    const int colB = blockIdx.x * 128;

    if (tid < 128) { rn[tid] = n1[rowB + tid]; colpart[tid] = 0.f; }
    else           { cn[tid - 128] = n2[colB + (tid - 128)]; }
    __syncthreads();

    v8f acc[8] = {};
    const int mRow = rowB + wave * 16 + l15;

    #pragma unroll
    for (int k0 = 0; k0 < 128; k0 += 32) {
        union { v16bf v; u32x4 p[2]; } af;
        const unsigned short* ap = Z1b + (size_t)mRow * 128 + k0 + hlf * 8;
        af.p[0] = *(const u32x4*)(ap);
        af.p[1] = *(const u32x4*)(ap + 16);
        #pragma unroll
        for (int t = 0; t < 8; ++t) {
            const unsigned short* bp =
                Z2b + (size_t)(colB + t * 16 + l15) * 128 + k0 + hlf * 8;
            union { v16bf v; u32x4 p[2]; } bf;
            bf.p[0] = *(const u32x4*)(bp);
            bf.p[1] = *(const u32x4*)(bp + 16);
            acc[t] = __builtin_amdgcn_wmma_f32_16x16x32_bf16(
                false, af.v, false, bf.v, (short)0, acc[t], false, false);
        }
    }

    // ---- exp + reductions (D layout: m = hlf*8+v, n = l15 within tile) ----
    float rsv[8];
    #pragma unroll
    for (int v = 0; v < 8; ++v) rsv[v] = 0.f;
    float csum[8];

    #pragma unroll
    for (int t = 0; t < 8; ++t) {
        float n2v = cn[t * 16 + l15];
        float ct = 0.f;
        #pragma unroll
        for (int v = 0; v < 8; ++v) {
            float n1v = rn[wave * 16 + (hlf << 3) + v];
            float e = __expf(acc[t][v] / (n1v * n2v) * (1.0f / TAUF));
            rsv[v] += e;
            ct += e;
        }
        ct += __shfl_xor(ct, 16, 32);   // lanes l, l^16 share column n
        csum[t] = ct;
    }

    #pragma unroll
    for (int off = 1; off < 16; off <<= 1) {
        #pragma unroll
        for (int v = 0; v < 8; ++v) rsv[v] += __shfl_xor(rsv[v], off, 32);
    }
    if (l15 == 0) {
        #pragma unroll
        for (int v = 0; v < 8; ++v)
            atomicAdd(&rowsum[rowB + wave * 16 + (hlf << 3) + v], rsv[v]);
    }
    if (hlf == 0) {
        #pragma unroll
        for (int t = 0; t < 8; ++t)
            atomicAdd(&colpart[t * 16 + l15], csum[t]);
    }
    __syncthreads();
    if (tid < 128) atomicAdd(&colsum[colB + tid], colpart[tid]);
}

// ---------------------------------------------------------------------------
__global__ __launch_bounds__(256) void loss_reduce(
    const float* __restrict__ d, const float* __restrict__ rs,
    const float* __restrict__ cs, float* __restrict__ out)
{
    __shared__ float s1[256], s2[256];
    int tid = threadIdx.x;
    float a1 = 0.f, a2 = 0.f;
    for (int i = tid; i < NN; i += 256) {
        float di = d[i];
        a1 += __logf(di / (rs[i] + EPSF) + EPSF);
        a2 += __logf(di / (cs[i] + EPSF) + EPSF);
    }
    s1[tid] = a1; s2[tid] = a2;
    __syncthreads();
    for (int off = 128; off > 0; off >>= 1) {
        if (tid < off) { s1[tid] += s1[tid + off]; s2[tid] += s2[tid + off]; }
        __syncthreads();
    }
    if (tid == 0)
        out[0] = 0.5f * (-(s1[0] / (float)NN) - (s2[0] / (float)NN));
}

// ---------------------------------------------------------------------------
extern "C" void kernel_launch(void* const* d_in, const int* in_sizes, int n_in,
                              void* d_out, int out_size, void* d_ws, size_t ws_size,
                              hipStream_t stream) {
    (void)in_sizes; (void)n_in; (void)out_size; (void)ws_size;

    const float* adj[3] = {(const float*)d_in[0], (const float*)d_in[1],
                           (const float*)d_in[2]};
    const float* feat   = (const float*)d_in[3];
    const float* W[3]   = {(const float*)d_in[4], (const float*)d_in[7],
                           (const float*)d_in[10]};
    const float* b[3]   = {(const float*)d_in[5], (const float*)d_in[8],
                           (const float*)d_in[11]};
    const float* a[3]   = {(const float*)d_in[6], (const float*)d_in[9],
                           (const float*)d_in[12]};
    const float* Wp1 = (const float*)d_in[13];
    const float* bp1 = (const float*)d_in[14];
    const float* Wp2 = (const float*)d_in[15];
    const float* bp2 = (const float*)d_in[16];

    float* ws = (float*)d_ws;
    const size_t NM = (size_t)NN * HH;           // 4 MB per [8192 x 128] f32
    float* XW[3] = {ws + 0 * NM, ws + 1 * NM, ws + 2 * NM};
    float* Hb[3] = {ws + 3 * NM, ws + 4 * NM, ws + 5 * NM};
    float* Tb    =  ws + 6 * NM;
    float* V[3]  = {ws + 7 * NM, ws + 8 * NM, ws + 9 * NM};
    float* nrm[3] = {ws + 10 * NM, ws + 10 * NM + NN, ws + 10 * NM + 2 * NN};
    float* dbuf = ws + 10 * NM + 3 * NN;
    float* rsb  = dbuf + NN;
    float* csb  = rsb + NN;
    unsigned short* Vbf[3];
    {
        unsigned short* base = (unsigned short*)(csb + NN);
        Vbf[0] = base; Vbf[1] = base + NM; Vbf[2] = base + 2 * NM;
    }

    dim3 blk(256);
    dim3 gM(NN / 128);

    // XW_i = feat @ W_i
    for (int i = 0; i < 3; ++i)
        wmma_gemm128<<<gM, blk, 0, stream>>>(feat, W[i], XW[i],
                                             FF, FF, nullptr, nullptr, EPI_NONE);
    // H_i = PReLU(adj_i @ XW_i + b_i)   (TDM streams 256 MB of adj per view)
    for (int i = 0; i < 3; ++i)
        wmma_gemm128<<<gM, blk, 0, stream>>>(adj[i], XW[i], Hb[i],
                                             NN, NN, b[i], a[i], EPI_PRELU);
    // V_i = ELU(H_i @ Wp1 + bp1) @ Wp2 + bp2
    for (int i = 0; i < 3; ++i) {
        wmma_gemm128<<<gM, blk, 0, stream>>>(Hb[i], Wp1, Tb,
                                             HH, HH, bp1, nullptr, EPI_ELU);
        wmma_gemm128<<<gM, blk, 0, stream>>>(Tb, Wp2, V[i],
                                             HH, HH, bp2, nullptr, EPI_BIAS);
    }
    for (int i = 0; i < 3; ++i) {
        norm128<<<NN / 8, blk, 0, stream>>>(V[i], nrm[i]);
        cvt_bf16<<<(int)(NM / (256 * 8)), blk, 0, stream>>>(V[i], Vbf[i]);
    }

    const int pairs[3][2] = {{0, 1}, {0, 2}, {1, 2}};
    for (int p = 0; p < 3; ++p) {
        int i = pairs[p][0], j = pairs[p][1];
        hipMemsetAsync(rsb, 0, NN * sizeof(float), stream);
        hipMemsetAsync(csb, 0, NN * sizeof(float), stream);
        diag128<<<NN / 8, blk, 0, stream>>>(V[i], V[j], nrm[i], nrm[j], dbuf);
        dim3 gT(NN / 128, NN / 128);
        nt_xent_tiles<<<gT, blk, 0, stream>>>(Vbf[i], Vbf[j], nrm[i], nrm[j],
                                              rsb, csb);
        loss_reduce<<<1, blk, 0, stream>>>(dbuf, rsb, csb, ((float*)d_out) + p);
    }
}
